// SingleKernelRetinalGlimpse_54709293416610
// MI455X (gfx1250) — matrix operand order repaired and checked by reading, not compiled, not run on hardware
//
#include <hip/hip_runtime.h>
#include <hip/hip_bf16.h>

typedef __attribute__((ext_vector_type(16))) __bf16 v16bf;
typedef __attribute__((ext_vector_type(8)))  float  v8f;

#define BATCH 32
#define NK    144
#define HW    1024
#define JTILE 128
#define MTILES 9   // 144 / 16

// round-to-nearest-even f32 -> bf16 (in low 16 bits)
__device__ __forceinline__ unsigned int bf16rne(float f) {
    unsigned int x = __float_as_uint(f);
    unsigned int r = x + 0x7FFFu + ((x >> 16) & 1u);
    return r >> 16;
}

// ---------------------------------------------------------------------------
// Prep: one wave per (b,k). Computes Gaussian rows kx (bf16, pre-swizzled into
// the CDNA5 16-bit A-matrix 16x32 fragment layout) and ky (f32, row-major),
// plus inv_norm = 1/(sum_kx * sum_ky). Also zeroes d_out (atomics target).
// kxA layout: [b][ichunk(32)][mtile(9)][lane(32)][16 bf16]  (32B per lane)
// ---------------------------------------------------------------------------
__global__ __launch_bounds__(256) void prep_kernel(
    const float* __restrict__ s_c, const float* __restrict__ s_z,
    const float* __restrict__ mu,  const float* __restrict__ sigma,
    unsigned short* __restrict__ kxA, float* __restrict__ ky,
    float* __restrict__ inv_norm, float* __restrict__ out)
{
    const int w    = threadIdx.x >> 5;
    const int lane = threadIdx.x & 31;
    const int p = blockIdx.x * 8 + w;      // 0..4607
    const int b = p / NK;
    const int k = p % NK;

    const float sz   = s_z[b];
    const float mux  = (s_c[b * 2 + 0] - mu[k * 2 + 0]) * sz;
    const float muy  = (s_c[b * 2 + 1] - mu[k * 2 + 1]) * sz;
    const float isig = 1.0f / (sigma[k] * sz);

    // A-fragment coordinates for in-chunk index il == lane (constant per lane)
    const int mt    = k >> 4;
    const int m     = k & 15;
    const int laneA = m + 16 * ((lane >> 3) & 1);
    const int vslot = ((lane >= 16) ? 4 : 0) + ((lane & 7) >> 1);
    const int slot  = lane & 1;
    size_t abase = (((size_t)(b * 32) * MTILES + mt) * 32 + laneA) * 16
                 + vslot * 2 + slot;                 // ushort units
    const size_t astride = (size_t)MTILES * 32 * 16; // per ichunk = 4608

    float sx = 0.0f, sy = 0.0f;
    #pragma unroll 4
    for (int t = 0; t < 32; ++t) {
        const int i = t * 32 + lane;
        const float xv = (float)i * (2.0f / 1023.0f) - 1.0f;
        float dx = (xv - mux) * isig;
        float kxv = __expf(-0.5f * dx * dx);
        sx += kxv;
        kxA[abase + (size_t)t * astride] = (unsigned short)bf16rne(kxv);
        float dy = (xv - muy) * isig;
        float kyv = __expf(-0.5f * dy * dy);
        sy += kyv;
        ky[(size_t)p * HW + i] = kyv;
    }
    #pragma unroll
    for (int s = 1; s < 32; s <<= 1) {
        sx += __shfl_xor(sx, s, 32);
        sy += __shfl_xor(sy, s, 32);
    }
    if (lane == 0) {
        inv_norm[p] = 1.0f / (sx * sy);
        out[p] = 0.0f;
    }
}

// ---------------------------------------------------------------------------
// Main: C[k,j] = sum_i kx[b,k,i] * U[b,i,j], fused with epilogue
// out[b,k] += inv_norm * sum_j C[k,j] * ky[b,k,j].
// Grid: (8 j-tiles, 32 batches), 256 threads = 8 waves.
// Wave w owns columns [j0 + 16w, +16) and all 9 M-tiles (144 kernels).
// ---------------------------------------------------------------------------
__global__ __launch_bounds__(256) void glimpse_wmma_kernel(
    const float* __restrict__ U, const v16bf* __restrict__ kxA,
    const float* __restrict__ ky, const float* __restrict__ inv_norm,
    float* __restrict__ out)
{
    __shared__ unsigned int lds_u[8 * 32 * 8]; // B fragments: [ntile][lane][vgpr dword]
    __shared__ float lds_out[NK];

    const int b   = blockIdx.y;
    const int j0  = blockIdx.x * JTILE;
    const int tid = threadIdx.x;
    const int w   = tid >> 5;
    const int lane = tid & 31;

    if (tid < NK) lds_out[tid] = 0.0f;

    v8f acc[MTILES];
    #pragma unroll
    for (int mt = 0; mt < MTILES; ++mt) acc[mt] = (v8f){0,0,0,0,0,0,0,0};

    const float* __restrict__ Ub = U + (size_t)b * HW * HW;

    for (int ic = 0; ic < 32; ++ic) {
        __syncthreads();
        const int i0 = ic * 32;
        // Stage 32x128 U tile -> bf16 in B-fragment layout.
        // Rows (2r, 2r+1) pack into one dword of VGPR v = 4*(il>>4) + ((il&7)>>1), lane-group g = (il>>3)&1.
        #pragma unroll
        for (int r = 0; r < 8; ++r) {
            const int idx = r * 256 + tid;      // 0..2047
            const int ilp = idx >> 7;           // row-pair 0..15
            const int jl  = idx & 127;
            const int il  = ilp * 2;
            const float u0 = Ub[(size_t)(i0 + il)     * HW + j0 + jl];
            const float u1 = Ub[(size_t)(i0 + il + 1) * HW + j0 + jl];
            const unsigned int pk = bf16rne(u0) | (bf16rne(u1) << 16);
            const int nt    = jl >> 4;
            const int g     = (il >> 3) & 1;
            const int laneb = (jl & 15) + 16 * g;
            const int v     = ((il >> 4) << 2) + ((il & 7) >> 1);
            lds_u[(nt * 32 + laneb) * 8 + v] = pk;
        }
        __syncthreads();

        const v16bf bfrag = *(const v16bf*)&lds_u[(w * 32 + lane) * 8];
        const v16bf* __restrict__ aptr =
            kxA + ((size_t)(b * 32 + ic) * MTILES) * 32 + lane;
        #pragma unroll
        for (int mt = 0; mt < MTILES; ++mt) {
            const v16bf afrag = aptr[(size_t)mt * 32];
            acc[mt] = __builtin_amdgcn_wmma_f32_16x16x32_bf16(
                false, afrag, false, bfrag, (short)0, acc[mt], false, false);
        }
    }
    __syncthreads();

    // Epilogue: C layout -> lane holds col n=lane&15; VGPR v holds row v+8*(lane>>4)
    const int half = lane >> 4;
    const int jcol = j0 + w * 16 + (lane & 15);
    #pragma unroll
    for (int mt = 0; mt < MTILES; ++mt) {
        #pragma unroll
        for (int v = 0; v < 8; ++v) {
            const int k = mt * 16 + 8 * half + v;
            const float kyv = ky[((size_t)b * NK + k) * HW + jcol];
            float t = acc[mt][v] * kyv;
            t += __shfl_xor(t, 1, 32);
            t += __shfl_xor(t, 2, 32);
            t += __shfl_xor(t, 4, 32);
            t += __shfl_xor(t, 8, 32);   // sum over 16 cols within each half
            if ((lane & 15) == v) atomicAdd(&lds_out[k], t);
        }
    }
    __syncthreads();
    if (tid < NK) {
        atomicAdd(&out[b * NK + tid], lds_out[tid] * inv_norm[b * NK + tid]);
    }
}

// ---------------------------------------------------------------------------
extern "C" void kernel_launch(void* const* d_in, const int* in_sizes, int n_in,
                              void* d_out, int out_size, void* d_ws, size_t ws_size,
                              hipStream_t stream) {
    const float* U     = (const float*)d_in[0];
    const float* s_c   = (const float*)d_in[1];
    const float* s_z   = (const float*)d_in[2];
    const float* mu    = (const float*)d_in[3];
    const float* sigma = (const float*)d_in[4];
    float* out = (float*)d_out;

    char* ws = (char*)d_ws;
    const size_t kxA_bytes = (size_t)BATCH * 32 * MTILES * 32 * 32; // 9,437,184
    const size_t ky_bytes  = (size_t)BATCH * NK * HW * 4;           // 18,874,368
    unsigned short* kxA = (unsigned short*)ws;
    float* ky           = (float*)(ws + kxA_bytes);
    float* inv_norm     = (float*)(ws + kxA_bytes + ky_bytes);

    // 4608 (b,k) pairs, 8 waves/block -> 576 blocks
    prep_kernel<<<576, 256, 0, stream>>>(s_c, s_z, mu, sigma, kxA, ky, inv_norm, out);

    dim3 grid(HW / JTILE, BATCH);  // (8, 32)
    glimpse_wmma_kernel<<<grid, 256, 0, stream>>>(
        U, (const v16bf*)kxA, ky, inv_norm, out);
}
